// MultiHeadAttentionBlock_59828894433684
// MI455X (gfx1250) — matrix operand order repaired and checked by reading, not compiled, not run on hardware
//
#include <hip/hip_runtime.h>
#include <hip/hip_bf16.h>

typedef __attribute__((ext_vector_type(16))) __bf16 v16bf;
typedef __attribute__((ext_vector_type(8)))  __bf16 v8bf;
typedef __attribute__((ext_vector_type(8)))  float  v8f;
typedef int v4i __attribute__((vector_size(16)));

#define AS1 __attribute__((address_space(1)))
#define AS3 __attribute__((address_space(3)))

#if defined(__AMDGCN__) && __has_builtin(__builtin_amdgcn_global_load_async_to_lds_b128)
#define HAVE_ASYNC_LDS 1
#else
#define HAVE_ASYNC_LDS 0
#endif

namespace {

constexpr int kDModel = 1024;
constexpr int kHeads  = 16;
constexpr int kDk     = 64;     // kDModel / kHeads
constexpr int kBatch  = 4;
constexpr int kSeq    = 2048;
constexpr int kRows   = kBatch * kSeq;   // 8192

union Frag16 { v16bf v; v8bf h[2]; };

__device__ __forceinline__ __bf16 bf(float f) { return (__bf16)f; }

__device__ __forceinline__ v8f wmma_bf16(v16bf a, v16bf b, v8f c) {
  // D(f32 16x16) = A(bf16 16x32) * B(bf16 32x16) + C
  return __builtin_amdgcn_wmma_f32_16x16x32_bf16(
      /*neg_a=*/false, a, /*neg_b=*/false, b,
      /*c_mod=*/(short)0, c, /*reuse_a=*/false, /*reuse_b=*/false);
}

#if HAVE_ASYNC_LDS
__device__ __forceinline__ void async_copy_b128(const void* g, void* l) {
  __builtin_amdgcn_global_load_async_to_lds_b128(
      (AS1 v4i*)g, (AS3 v4i*)l, /*offset=*/0, /*cpol=*/0);
}
__device__ __forceinline__ void wait_async0() {
#if __has_builtin(__builtin_amdgcn_s_wait_asynccnt)
  __builtin_amdgcn_s_wait_asynccnt(0);
#else
  asm volatile("s_wait_asynccnt 0x0" ::: "memory");
#endif
}
#endif

// ---------------------------------------------------------------------------
// GEMM: C[M,N] = A[M,K] * B[K,N] + bias[N]
// M=8192, N=K=1024. Block = 256 thr (8 waves), tile 128x128, k-step 32.
// A_BF16:   A is bf16 (ctx) vs fp32 (raw activations).
// PERM_OUT: write bf16 head-split [B,H,T,Dk]; else write fp32 row-major.
// ---------------------------------------------------------------------------
template<bool A_BF16, bool PERM_OUT>
__global__ __launch_bounds__(256) void gemm_wmma(
    const void* __restrict__ Ap, const float* __restrict__ Bw,
    const float* __restrict__ bias, void* __restrict__ Cp)
{
  constexpr int N = kDModel, K = kDModel;
  constexpr int LDA = 40;   // padded bf16 row stride (80B: 16B-aligned, bank-skewed)
  constexpr int LDB = 40;
  __shared__ __bf16 As[128 * LDA];           // [m][k] 128x32
  __shared__ __bf16 Bs[128 * LDB];           // transposed: [n][k] 128x32

  const int m0 = blockIdx.x * 128;
  const int n0 = blockIdx.y * 128;
  const int tid  = threadIdx.x;
  const int wave = tid >> 5, lane = tid & 31;
  const int wm = wave & 3, wn = wave >> 2;   // wave tile: rows wm*32, cols wn*64
  const int lh = lane >> 4, ln = lane & 15;

  v8f acc[2][4];
  for (int mi = 0; mi < 2; ++mi)
    for (int ni = 0; ni < 4; ++ni)
      for (int r = 0; r < 8; ++r) acc[mi][ni][r] = 0.0f;

  for (int k0 = 0; k0 < K; k0 += 32) {
    __syncthreads();
    if (A_BF16) {
      const __bf16* A = (const __bf16*)Ap;
#if HAVE_ASYNC_LDS
      #pragma unroll
      for (int i = 0; i < 2; ++i) {
        int slot = tid + 256 * i;            // 512 slots of 8 bf16
        int r = slot >> 2, c8 = slot & 3;
        async_copy_b128(A + (size_t)(m0 + r) * K + k0 + c8 * 8,
                        &As[r * LDA + c8 * 8]);
      }
#else
      #pragma unroll
      for (int i = 0; i < 2; ++i) {
        int slot = tid + 256 * i;            // 512 slots of 8 bf16
        int r = slot >> 2, c8 = slot & 3;
        uint4 u = *(const uint4*)(A + (size_t)(m0 + r) * K + k0 + c8 * 8);
        *(uint4*)&As[r * LDA + c8 * 8] = u;
      }
#endif
    } else {
      const float* A = (const float*)Ap;
      #pragma unroll
      for (int i = 0; i < 4; ++i) {
        int slot = tid + 256 * i;            // 1024 slots of float4
        int r = slot >> 3, c4 = slot & 7;
        float4 f = *(const float4*)(A + (size_t)(m0 + r) * K + k0 + c4 * 4);
        __bf16* d = &As[r * LDA + c4 * 4];
        d[0] = bf(f.x); d[1] = bf(f.y); d[2] = bf(f.z); d[3] = bf(f.w);
      }
    }
    // B tile: 32 k-rows x 128 n-cols, stored transposed [n][k]
    #pragma unroll
    for (int i = 0; i < 4; ++i) {
      int slot = tid + 256 * i;              // 1024 slots of float4
      int kr = slot >> 5, nf = slot & 31;
      float4 f = *(const float4*)(Bw + (size_t)(k0 + kr) * N + n0 + nf * 4);
      Bs[(nf * 4 + 0) * LDB + kr] = bf(f.x);
      Bs[(nf * 4 + 1) * LDB + kr] = bf(f.y);
      Bs[(nf * 4 + 2) * LDB + kr] = bf(f.z);
      Bs[(nf * 4 + 3) * LDB + kr] = bf(f.w);
    }
#if HAVE_ASYNC_LDS
    if (A_BF16) wait_async0();
#endif
    __syncthreads();

    // prefetch next tiles into cache while this one computes
    if (k0 + 32 < K) {
      const char* Abase = (const char*)Ap;
      const size_t esz = A_BF16 ? 2u : 4u;
      __builtin_prefetch(
          Abase + ((size_t)(m0 + (tid >> 1)) * K + (k0 + 32) + (tid & 1) * 16) * esz, 0, 3);
      __builtin_prefetch(
          (const char*)Bw + (((size_t)(k0 + 32 + (tid >> 4)) * N) + n0 + (tid & 15) * 8) * 4, 0, 3);
    }

    // A fragments: lane ln = row, elems 0-7 = K lh*8.., elems 8-15 = K 16+lh*8..
    Frag16 af[2];
    #pragma unroll
    for (int mi = 0; mi < 2; ++mi) {
      const __bf16* base = &As[(wm * 32 + mi * 16 + ln) * LDA];
      af[mi].h[0] = *(const v8bf*)(base + lh * 8);
      af[mi].h[1] = *(const v8bf*)(base + 16 + lh * 8);
    }
    // B fragments: lane ln = col, elems 0-7 = K lh*8.., elems 8-15 = K 16+lh*8..
    #pragma unroll
    for (int ni = 0; ni < 4; ++ni) {
      Frag16 bfr;
      const __bf16* base = &Bs[(wn * 64 + ni * 16 + ln) * LDB + lh * 8];
      bfr.h[0] = *(const v8bf*)(base);
      bfr.h[1] = *(const v8bf*)(base + 16);
      #pragma unroll
      for (int mi = 0; mi < 2; ++mi)
        acc[mi][ni] = wmma_bf16(af[mi].v, bfr.v, acc[mi][ni]);
    }
  }

  // Epilogue: C layout -> lane holds (M = r + 8*lh, N = ln)
  #pragma unroll
  for (int ni = 0; ni < 4; ++ni) {
    const int col = n0 + wn * 64 + ni * 16 + ln;
    const float bv = bias[col];
    #pragma unroll
    for (int mi = 0; mi < 2; ++mi) {
      #pragma unroll
      for (int r = 0; r < 8; ++r) {
        const int row = m0 + wm * 32 + mi * 16 + r + 8 * lh;
        const float v = acc[mi][ni][r] + bv;
        if (PERM_OUT) {
          const int t = row & (kSeq - 1);
          const int b = row >> 11;            // / kSeq
          const int h = col >> 6;             // / kDk
          const int d = col & (kDk - 1);
          ((__bf16*)Cp)[((size_t)(b * kHeads + h) * kSeq + t) * kDk + d] = bf(v);
        } else {
          ((float*)Cp)[(size_t)row * N + col] = v;
        }
      }
    }
  }
}

// ---------------------------------------------------------------------------
// Flash attention: one block per (b,h, 64-query tile). 128 thr = 4 waves,
// each wave owns 16 query rows; K/V streamed in 64-key tiles via LDS.
// ---------------------------------------------------------------------------
__global__ __launch_bounds__(128) void attn_wmma(
    const void* __restrict__ Qp, const void* __restrict__ Kp,
    const void* __restrict__ Vp, void* __restrict__ Cp)
{
  constexpr int LDT = 72;                    // padded bf16 row stride (144B)
  __shared__ __bf16 Ks[64 * LDT];            // [s][d]  (already B-layout for QK^T)
  __shared__ __bf16 Vs[64 * LDT];            // transposed [d][s] (B-layout for P*V)
  __shared__ __bf16 Ps[4 * 16 * LDT];        // per-wave P pane [m][s]

  const int qt = blockIdx.x & (kSeq / 64 - 1);   // query tile 0..31
  const int bh = blockIdx.x >> 5;                // 0..63 = b*16+h
  const __bf16* Q  = (const __bf16*)Qp + (size_t)bh * kSeq * kDk;
  const __bf16* Kg = (const __bf16*)Kp + (size_t)bh * kSeq * kDk;
  const __bf16* Vg = (const __bf16*)Vp + (size_t)bh * kSeq * kDk;

  const int tid  = threadIdx.x;
  const int wave = tid >> 5, lane = tid & 31;
  const int lh = lane >> 4, ln = lane & 15;

  // Q fragments (held in registers the whole time): 16 rows x 64 d
  Frag16 qa[2];
  {
    const __bf16* qrow = Q + (size_t)(qt * 64 + wave * 16 + ln) * kDk;
    #pragma unroll
    for (int c = 0; c < 2; ++c) {
      qa[c].h[0] = *(const v8bf*)(qrow + c * 32 + lh * 8);
      qa[c].h[1] = *(const v8bf*)(qrow + c * 32 + 16 + lh * 8);
    }
  }

  float m_i[8], l_i[8];
  v8f o[4];
  #pragma unroll
  for (int r = 0; r < 8; ++r) { m_i[r] = -1e30f; l_i[r] = 0.0f; }
  for (int nt = 0; nt < 4; ++nt)
    for (int r = 0; r < 8; ++r) o[nt][r] = 0.0f;

  for (int s0 = 0; s0 < kSeq; s0 += 64) {
    __syncthreads();
    // K tile: 64 keys x 64 d, natural layout (pure 16B copies -> async path)
#if HAVE_ASYNC_LDS
    #pragma unroll
    for (int i = 0; i < 4; ++i) {
      int slot = tid + 128 * i;              // 512 slots of 8 bf16
      int r = slot >> 3, c8 = slot & 7;
      async_copy_b128(Kg + (size_t)(s0 + r) * kDk + c8 * 8,
                      &Ks[r * LDT + c8 * 8]);
    }
#else
    #pragma unroll
    for (int i = 0; i < 4; ++i) {
      int slot = tid + 128 * i;              // 512 slots of 8 bf16
      int r = slot >> 3, c8 = slot & 7;
      uint4 u = *(const uint4*)(Kg + (size_t)(s0 + r) * kDk + c8 * 8);
      *(uint4*)&Ks[r * LDT + c8 * 8] = u;
    }
#endif
    // V tile transposed into [d][s] (needs swizzle -> VGPR path)
    #pragma unroll
    for (int i = 0; i < 16; ++i) {
      int slot = tid + 128 * i;              // 2048 slots of 2 bf16
      int r = slot >> 5, dp = slot & 31;
      unsigned u = *(const unsigned*)(Vg + (size_t)(s0 + r) * kDk + dp * 2);
      __bf16 two[2];
      *(unsigned*)two = u;
      Vs[(dp * 2 + 0) * LDT + r] = two[0];
      Vs[(dp * 2 + 1) * LDT + r] = two[1];
    }
#if HAVE_ASYNC_LDS
    wait_async0();
#endif
    __syncthreads();

    // prefetch next K/V tiles (128 thr x 64B = 8KB each)
    if (s0 + 64 < kSeq) {
      __builtin_prefetch((const void*)(Kg + (size_t)(s0 + 64) * kDk + tid * 32), 0, 3);
      __builtin_prefetch((const void*)(Vg + (size_t)(s0 + 64) * kDk + tid * 32), 0, 3);
    }

    // S = (Q K^T) / sqrt(Dk): 4 score tiles of 16x16, K-dim = d (2 chunks of 32)
    v8f st[4];
    #pragma unroll
    for (int t4 = 0; t4 < 4; ++t4) {
      v8f s;
      for (int r = 0; r < 8; ++r) s[r] = 0.0f;
      #pragma unroll
      for (int c = 0; c < 2; ++c) {
        Frag16 kf;
        const __bf16* base = &Ks[(t4 * 16 + ln) * LDT + c * 32 + lh * 8];
        kf.h[0] = *(const v8bf*)(base);
        kf.h[1] = *(const v8bf*)(base + 16);
        s = wmma_bf16(qa[c].v, kf.v, s);
      }
      for (int r = 0; r < 8; ++r) s[r] *= 0.125f;   // 1/sqrt(64)
      st[t4] = s;
    }

    // online softmax: rows live at (r + 8*lh), cols across 16 lanes
    float m_new[8], alpha[8];
    #pragma unroll
    for (int r = 0; r < 8; ++r) {
      float mx = fmaxf(fmaxf(st[0][r], st[1][r]), fmaxf(st[2][r], st[3][r]));
      #pragma unroll
      for (int off = 1; off < 16; off <<= 1)
        mx = fmaxf(mx, __shfl_xor(mx, off, 32));
      m_new[r] = fmaxf(m_i[r], mx);
      alpha[r] = __expf(m_i[r] - m_new[r]);
      l_i[r] *= alpha[r];
      m_i[r] = m_new[r];
    }
    #pragma unroll
    for (int nt = 0; nt < 4; ++nt)
      for (int r = 0; r < 8; ++r) o[nt][r] *= alpha[r];

    // P = exp(S - m): round-trip through per-wave LDS pane for C->A relayout
    __bf16* Pw = &Ps[wave * 16 * LDT];
    #pragma unroll
    for (int r = 0; r < 8; ++r) {
      float rs = 0.0f;
      const int rowm = r + 8 * lh;
      #pragma unroll
      for (int t4 = 0; t4 < 4; ++t4) {
        float p = __expf(st[t4][r] - m_new[r]);
        rs += p;
        Pw[rowm * LDT + t4 * 16 + ln] = bf(p);
      }
      #pragma unroll
      for (int off = 1; off < 16; off <<= 1)
        rs += __shfl_xor(rs, off, 32);
      l_i[r] += rs;
    }

    // O += P(16x64) * V(64x64); LDS is in-order within a wave (DScnt)
    Frag16 pa[2];
    #pragma unroll
    for (int c = 0; c < 2; ++c) {
      const __bf16* pb = Pw + ln * LDT + c * 32;
      pa[c].h[0] = *(const v8bf*)(pb + lh * 8);
      pa[c].h[1] = *(const v8bf*)(pb + 16 + lh * 8);
    }
    #pragma unroll
    for (int nt = 0; nt < 4; ++nt) {
      #pragma unroll
      for (int c = 0; c < 2; ++c) {
        Frag16 vf;
        const __bf16* base = &Vs[(nt * 16 + ln) * LDT + c * 32 + lh * 8];
        vf.h[0] = *(const v8bf*)(base);
        vf.h[1] = *(const v8bf*)(base + 16);
        o[nt] = wmma_bf16(pa[c].v, vf.v, o[nt]);
      }
    }
  }

  // normalize + write ctx as [B, T, D_MODEL] bf16 (merge heads)
  const int b = bh >> 4, h = bh & (kHeads - 1);
  __bf16* C = (__bf16*)Cp;
  #pragma unroll
  for (int r = 0; r < 8; ++r) {
    const float inv = 1.0f / l_i[r];
    const int t = qt * 64 + wave * 16 + r + 8 * lh;
    #pragma unroll
    for (int nt = 0; nt < 4; ++nt) {
      const int col = h * kDk + nt * 16 + ln;
      C[(size_t)(b * kSeq + t) * kDModel + col] = bf(o[nt][r] * inv);
    }
  }
}

} // namespace

extern "C" void kernel_launch(void* const* d_in, const int* in_sizes, int n_in,
                              void* d_out, int out_size, void* d_ws, size_t ws_size,
                              hipStream_t stream) {
  const float* query = (const float*)d_in[0];
  const float* key   = (const float*)d_in[1];
  const float* value = (const float*)d_in[2];
  const float* Wq    = (const float*)d_in[3];
  const float* bq    = (const float*)d_in[4];
  const float* Wk    = (const float*)d_in[5];
  const float* bk    = (const float*)d_in[6];
  const float* Wv    = (const float*)d_in[7];
  const float* bv    = (const float*)d_in[8];
  const float* Wo    = (const float*)d_in[9];
  const float* bo    = (const float*)d_in[10];

  // workspace: Q/K/V in [B,H,T,Dk] bf16 (16 MB each), ctx [B,T,D] bf16 (16 MB)
  char* ws = (char*)d_ws;
  void* Qws = (void*)(ws + (size_t)0);
  void* Kws = (void*)(ws + (size_t)16 * 1024 * 1024);
  void* Vws = (void*)(ws + (size_t)32 * 1024 * 1024);
  void* Cws = (void*)(ws + (size_t)48 * 1024 * 1024);

  const dim3 ggemm(kRows / 128, kDModel / 128);   // 64 x 8 blocks

  gemm_wmma<false, true><<<ggemm, 256, 0, stream>>>((const void*)query, Wq, bq, Qws);
  gemm_wmma<false, true><<<ggemm, 256, 0, stream>>>((const void*)key,   Wk, bk, Kws);
  gemm_wmma<false, true><<<ggemm, 256, 0, stream>>>((const void*)value, Wv, bv, Vws);

  attn_wmma<<<dim3(kBatch * kHeads * (kSeq / 64)), 128, 0, stream>>>(Qws, Kws, Vws, Cws);

  gemm_wmma<true, false><<<ggemm, 256, 0, stream>>>((const void*)Cws, Wo, bo, d_out);
}
